// observed_RNN_47820165873916
// MI455X (gfx1250) — matrix operand order repaired and checked by reading, not compiled, not run on hardware
//
#include <hip/hip_runtime.h>

typedef __attribute__((ext_vector_type(16))) _Float16 v16h;
typedef __attribute__((ext_vector_type(8)))  _Float16 v8h;
typedef __attribute__((ext_vector_type(8)))  float    v8f;

#define N_TRIALS 32
#define T_STEPS  2048
#define IN_DIM   64
#define NN       256
#define DT_C     0.1f

__device__ __forceinline__ float sigmoidf_fast(float x) {
    return 1.0f / (1.0f + __expf(-x));
}

// A-fragment (16x32 f16, activations). Lane l: m = l&15, g = l>>4.
// e=0..7  -> K = k0 + 8*g + e           (contiguous 16B)
// e=8..15 -> K = k0 + 16 + 8*g + (e-8)  (contiguous 16B)
__device__ __forceinline__ v16h load_a_frag(const _Float16* base, int row,
                                            int stride, int k0, int g) {
    const v8h lo = *(const v8h*)(base + row * stride + k0 + 8 * g);
    const v8h hi = *(const v8h*)(base + row * stride + k0 + 16 + 8 * g);
    v16h a;
#pragma unroll
    for (int e = 0; e < 8; ++e) { a[e] = lo[e]; a[8 + e] = hi[e]; }
    return a;
}

// B-fragment (32x16 f16, weights): lane l holds column n0 + (l&15);
// e=0..7 -> K = k0 + 8*g + e, e=8..15 -> K = k0 + 16 + 8*g + (e-8).
__device__ __forceinline__ v16h load_b_frag(const _Float16* base, int krow,
                                            int stride, int n0) {
    const int ln = krow & 31;
    const int k0 = krow - ln;
    const int gg = ln >> 4;
    const int n  = n0 + (ln & 15);
    v16h b;
#pragma unroll
    for (int e = 0; e < 8; ++e) {
        b[e]     = base[(k0 + 8 * gg + e) * stride + n];
        b[8 + e] = base[(k0 + 16 + 8 * gg + e) * stride + n];
    }
    return b;
}

__global__ void __launch_bounds__(256)
observed_rnn_kernel(const float* __restrict__ inp,   // (32, 2048, 64)
                    const float* __restrict__ J,     // (256, 256)
                    const float* __restrict__ Bm,    // (256, 64)
                    const float* __restrict__ W,     // (256, 256)
                    float* __restrict__ out) {       // x_seq | out_seq
    // ---- LDS-resident weights + activation ping-pong: 306 KB of 320 KB WGP LDS
    __shared__ __align__(64) _Float16 Jt[NN * NN];        // J^T  : 128 KB
    __shared__ __align__(64) _Float16 Wt[NN * NN];        // W    : 128 KB
    __shared__ __align__(64) _Float16 Bt[IN_DIM * NN];    // B^T  :  32 KB
    __shared__ __align__(64) _Float16 Sbuf[2][16 * NN];   // sigm :  16 KB
    __shared__ __align__(64) _Float16 Ubuf[16 * IN_DIM];  // u(t) :   2 KB

    const int tid  = threadIdx.x;
    const int tr0  = blockIdx.x * 16;          // this WG owns 16 trials
    const int lane = tid & 31;
    const int wave = tid >> 5;                 // 8 waves
    const int n0   = wave * 32;                // wave owns N columns [n0, n0+32)
    const int n1   = n0 + 16;
    const int nl   = lane & 15;
    const int g    = lane >> 4;
    const int mrow = lane & 15;

    float* xseq = out;
    float* oseq = out + (size_t)N_TRIALS * (T_STEPS + 1) * NN;

    // ---- Prologue: load + convert weights into LDS (one time; amortized over 2048 steps)
    for (int i = tid; i < NN * NN; i += 256)
        Wt[i] = (_Float16)W[i];                          // W used un-transposed
    for (int i = tid; i < NN * NN; i += 256) {           // Jt[k][n] = J[n][k]
        int n = i >> 8, k = i & 255;
        Jt[k * NN + n] = (_Float16)J[i];
    }
    for (int i = tid; i < NN * IN_DIM; i += 256) {       // Bt[k][n] = B[n][k]
        int n = i >> 6, k = i & 63;
        Bt[k * NN + n] = (_Float16)Bm[i];
    }
    for (int i = tid; i < 16 * NN; i += 256)             // sigmoid(0) = 0.5
        Sbuf[0][i] = (_Float16)0.5f;
    for (int i = tid; i < 16 * NN; i += 256) {           // t=0 rows are zeros
        int m = i >> 8, n = i & 255;
        size_t off = ((size_t)(tr0 + m) * (T_STEPS + 1)) * NN + n;
        xseq[off] = 0.0f;
        oseq[off] = 0.0f;
    }

    // Each thread stages one float4 of u(t): row m = tid>>4, cols d0..d0+3.
    const int um  = tid >> 4;
    const int ud0 = (tid & 15) * 4;
    const float* ubase = inp + (size_t)(tr0 + um) * T_STEPS * IN_DIM + ud0;

    // Preload u(0) into registers before entering the serial loop.
    float4 ureg = *(const float4*)(ubase);

    __syncthreads();

    // ---- State x lives in accumulator-layout registers (2 tiles x v8f per wave)
    v8f xf0, xf1;
#pragma unroll
    for (int r = 0; r < 8; ++r) { xf0[r] = 0.0f; xf1[r] = 0.0f; }

    for (int t = 0; t < T_STEPS; ++t) {
        // -- Phase A: commit prefetched u(t) into LDS as f16
        {
            _Float16* dst = &Ubuf[um * IN_DIM + ud0];
            dst[0] = (_Float16)ureg.x; dst[1] = (_Float16)ureg.y;
            dst[2] = (_Float16)ureg.z; dst[3] = (_Float16)ureg.w;
        }
        __syncthreads();

        // Prefetch u(t+1) now; its wait lands in next step's Phase A, fully
        // hidden behind this step's ~36 WMMAs. Clamp to stay in bounds.
        {
            const int tn = (t + 1 < T_STEPS) ? (t + 1) : (T_STEPS - 1);
            ureg = *(const float4*)(ubase + (size_t)tn * IN_DIM);
        }

        const _Float16* Scur = Sbuf[t & 1];
        _Float16* Snxt = Sbuf[(t + 1) & 1];

        // -- Phase B: acc = sigmoid(x) @ J^T + u @ B^T ; leak-update x; stage sigmoid(x')
        {
            v8f acc0 = {}, acc1 = {};
#pragma unroll
            for (int kb = 0; kb < 2; ++kb) {             // input term, K = 64
                v16h a  = load_a_frag(Ubuf, mrow, IN_DIM, kb * 32, g);
                v16h b0 = load_b_frag(Bt, kb * 32 + lane, NN, n0);
                v16h b1 = load_b_frag(Bt, kb * 32 + lane, NN, n1);
                acc0 = __builtin_amdgcn_wmma_f32_16x16x32_f16(
                    false, a, false, b0, (short)0, acc0, false, false);
                acc1 = __builtin_amdgcn_wmma_f32_16x16x32_f16(
                    false, a, false, b1, (short)0, acc1, false, false);
            }
#pragma unroll
            for (int kb = 0; kb < 8; ++kb) {             // recurrent term, K = 256
                v16h a  = load_a_frag(Scur, mrow, NN, kb * 32, g);
                v16h b0 = load_b_frag(Jt, kb * 32 + lane, NN, n0);
                v16h b1 = load_b_frag(Jt, kb * 32 + lane, NN, n1);
                acc0 = __builtin_amdgcn_wmma_f32_16x16x32_f16(
                    false, a, false, b0, (short)0, acc0, false, false);
                acc1 = __builtin_amdgcn_wmma_f32_16x16x32_f16(
                    false, a, false, b1, (short)0, acc1, false, false);
            }
#pragma unroll
            for (int r = 0; r < 8; ++r) {
                const int m = r + 8 * g;
                const size_t off =
                    ((size_t)(tr0 + m) * (T_STEPS + 1) + (t + 1)) * NN + nl;
                const float xn0 = (1.0f - DT_C) * xf0[r] + DT_C * acc0[r];
                const float xn1 = (1.0f - DT_C) * xf1[r] + DT_C * acc1[r];
                xf0[r] = xn0;
                xf1[r] = xn1;
                xseq[off + n0] = xn0;
                xseq[off + n1] = xn1;
                Snxt[m * NN + n0 + nl] = (_Float16)sigmoidf_fast(xn0);
                Snxt[m * NN + n1 + nl] = (_Float16)sigmoidf_fast(xn1);
            }
        }
        __syncthreads();

        // -- Phase C: out = sigmoid(x') @ W
        {
            v8f acc0 = {}, acc1 = {};
#pragma unroll
            for (int kb = 0; kb < 8; ++kb) {
                v16h a  = load_a_frag(Snxt, mrow, NN, kb * 32, g);
                v16h b0 = load_b_frag(Wt, kb * 32 + lane, NN, n0);
                v16h b1 = load_b_frag(Wt, kb * 32 + lane, NN, n1);
                acc0 = __builtin_amdgcn_wmma_f32_16x16x32_f16(
                    false, a, false, b0, (short)0, acc0, false, false);
                acc1 = __builtin_amdgcn_wmma_f32_16x16x32_f16(
                    false, a, false, b1, (short)0, acc1, false, false);
            }
#pragma unroll
            for (int r = 0; r < 8; ++r) {
                const int m = r + 8 * g;
                const size_t off =
                    ((size_t)(tr0 + m) * (T_STEPS + 1) + (t + 1)) * NN + nl;
                oseq[off + n0] = acc0[r];
                oseq[off + n1] = acc1[r];
            }
        }
        // No barrier needed here: the B<->C barrier already ordered all Ubuf/S
        // readers before any wave loops back to Phase A.
    }
}

extern "C" void kernel_launch(void* const* d_in, const int* in_sizes, int n_in,
                              void* d_out, int out_size, void* d_ws, size_t ws_size,
                              hipStream_t stream) {
    (void)in_sizes; (void)n_in; (void)out_size; (void)d_ws; (void)ws_size;
    const float* inp = (const float*)d_in[0];  // (32, 2048, 64)
    const float* J   = (const float*)d_in[1];  // (256, 256)
    const float* Bm  = (const float*)d_in[2];  // (256, 64)
    const float* W   = (const float*)d_in[3];  // (256, 256)
    float* out = (float*)d_out;

    observed_rnn_kernel<<<dim3(2), dim3(256), 0, stream>>>(inp, J, Bm, W, out);
}
